// Transformer_6863357739127
// MI455X (gfx1250) — matrix operand order repaired and checked by reading, not compiled, not run on hardware
//
#include <hip/hip_runtime.h>
#include <hip/hip_bf16.h>

typedef __attribute__((ext_vector_type(16))) __bf16 v16bf;
typedef __attribute__((ext_vector_type(8)))  float  v8f;
typedef __attribute__((ext_vector_type(4)))  unsigned int v4u;
typedef __attribute__((ext_vector_type(8)))  int v8i_;
typedef __attribute__((ext_vector_type(4)))  int v4i_;

#define DIM     1024
#define HIDDEN  4096
#define NTOK    1024
#define TSTEPS  10

#define LSTRIDE 40                 // bf16 elements per LDS tile row (80 B)
#define TILE_BF16 (128 * LSTRIDE)  // one 128x32(+pad) tile
#define TILE_BYTES (TILE_BF16 * 2)

union FragCvt { uint4 u[2]; v16bf v; };

// ---------------------------------------------------------------------------
// TDM: async-load a 2D tile [128 rows x 32 bf16] (row stride K elements) from
// global into LDS at byte offset lds_off, padding +16B after every 64B row so
// the LDS layout is exactly our stride-40-bf16 (80B) tile.
// D# layout per CDNA5 ISA 8.3/8.4: group0 {count=1, lds_addr, global_addr,
// type=2}; group1 {data_size=2B, pad_en, pad_interval=64B, pad_amount=16B,
// tensor_dim0=K, tensor_dim1=M, tile_dim0=32, tile_dim1=128, dim0_stride=K}.
// ---------------------------------------------------------------------------
__device__ __forceinline__ void tdm_load_a_tile(unsigned int lds_off,
                                                unsigned long long ga,
                                                int M, int K) {
  v4u g0 = { 1u,                                   // count=1 (user descriptor)
             lds_off,                              // lds_addr [63:32]
             (unsigned int)ga,                     // global_addr low
             (unsigned int)(ga >> 32) | 0x80000000u }; // addr hi | type=2
  v8i_ g1 = { (int)0x06D10000u,  // data_size=1(2B)<<16 | pad_en<<20 | 3<<22 | 3<<25
              (int)((K & 0xffff) << 16),                       // tensor_dim0 lo
              (int)(((unsigned)K >> 16) | ((M & 0xffff) << 16)),// dim0 hi | dim1 lo
              (int)(((unsigned)M >> 16) | (32u << 16)),        // dim1 hi | tile_dim0=32
              128,                                             // tile_dim1=128
              K,                                               // dim0_stride lo
              0, 0 };
  v4i_ gz = { 0, 0, 0, 0 };
#if defined(__clang_major__) && (__clang_major__ >= 23)
  v8i_ gz8 = { 0, 0, 0, 0, 0, 0, 0, 0 };
  __builtin_amdgcn_tensor_load_to_lds(g0, g1, gz, gz, gz8, 0);
#else
  __builtin_amdgcn_tensor_load_to_lds(g0, g1, gz, gz, 0);
#endif
}

// ---------------------------------------------------------------------------
// Tiled bf16 WMMA GEMM:  C[M,N](f32) = A[M,K](bf16) @ B[K,N](f32->bf16) (+C)
// Block tile 128x128, 8 waves (wave32), each wave 32x64 via 2x4 WMMA frags.
// Double-buffered software pipeline: A tile i+1 streams in via the Tensor
// Data Mover (TENSORcnt), B tile i+1 via global loads + packed bf16x2 LDS
// stores, both overlapped with the WMMAs of tile i. s_wait_tensorcnt(1)
// retires only the older in-flight TDM op (same-wave tensor ops complete
// in order).
// ---------------------------------------------------------------------------
__global__ __launch_bounds__(256) void gemm_bf16(
    const __bf16* __restrict__ A, const float* __restrict__ B,
    float* __restrict__ C, int M, int N, int K, int accumulate)
{
  __shared__ __align__(16) __bf16 Alds[2][TILE_BF16];   // [row][k], stride 40
  __shared__ __align__(16) __bf16 Blds[2][TILE_BF16];   // transposed: [n][k]

  const int tid  = threadIdx.x;
  const int lane = tid & 31;
  const int wave = tid >> 5;
  const int wm   = wave >> 1;               // 0..3
  const int wn   = wave & 1;                // 0..1
  const int rowBase = blockIdx.y * 128 + wm * 32;
  const int colBase = blockIdx.x * 128 + wn * 64;
  const int cB0     = blockIdx.x * 128;
  const int lrow = lane & 15;
  const int lhi  = lane >> 4;

  const unsigned int ldsA0 = (unsigned int)(uintptr_t)&Alds[0][0];
  const __bf16* aTileBase = A + (size_t)(blockIdx.y * 128) * K;

  const int bk2 = (tid >> 4) * 2;       // 0,2,..,30 : this thread's K-row pair
  const int bn0 = (tid & 15) * 8;       // 0..120    : 8 consecutive columns

  // stage K-tile kt into LDS buffer buf (A via TDM, B via global+LDS stores)
  auto stage = [&](int kt, int buf) {
    if (wave == 0) {
      unsigned long long ga =
          (unsigned long long)(uintptr_t)(aTileBase + kt * 32);
      tdm_load_a_tile(ldsA0 + (unsigned)buf * TILE_BYTES, ga, M, K);
    }
    const float* s0 = B + (size_t)(kt * 32 + bk2) * N + cB0 + bn0;
    const float* s1 = s0 + N;
    float4 a0 = ((const float4*)s0)[0], a1 = ((const float4*)s0)[1];
    float4 b0 = ((const float4*)s1)[0], b1 = ((const float4*)s1)[1];
    float lo[8] = { a0.x, a0.y, a0.z, a0.w, a1.x, a1.y, a1.z, a1.w };
    float hi[8] = { b0.x, b0.y, b0.z, b0.w, b1.x, b1.y, b1.z, b1.w };
    #pragma unroll
    for (int i = 0; i < 8; ++i) {
      union { __bf16 h[2]; unsigned int u; } p;
      p.h[0] = (__bf16)lo[i];
      p.h[1] = (__bf16)hi[i];
      *(unsigned int*)&Blds[buf][(bn0 + i) * LSTRIDE + bk2] = p.u; // 4B aligned
    }
    if ((kt + 1) * 32 < K) {   // CDNA5 global_prefetch_b8 for the tile after
      __builtin_prefetch(aTileBase + (size_t)(tid >> 1) * K + (kt + 1) * 32 + (tid & 1) * 16, 0, 1);
      __builtin_prefetch(B + (size_t)((kt + 1) * 32 + bk2) * N + cB0 + bn0, 0, 1);
    }
  };

  v8f acc[2][4];
  #pragma unroll
  for (int mi = 0; mi < 2; ++mi)
    #pragma unroll
    for (int ni = 0; ni < 4; ++ni) {
      if (accumulate) {
        #pragma unroll
        for (int i = 0; i < 8; ++i)
          acc[mi][ni][i] = C[(size_t)(rowBase + mi*16 + lhi*8 + i) * N + colBase + ni*16 + lrow];
      } else {
        #pragma unroll
        for (int i = 0; i < 8; ++i) acc[mi][ni][i] = 0.0f;
      }
    }

  const int NT = K >> 5;      // K-tiles of 32
  stage(0, 0);                // prologue

  for (int it = 0; it < NT; ++it) {
    const int cur = it & 1;
    const bool hasNext = (it + 1) < NT;
    if (hasNext) stage(it + 1, cur ^ 1);          // overlap DMA with compute
    if (wave == 0) {
      if (hasNext) __builtin_amdgcn_s_wait_tensorcnt(1);  // tile `cur` done
      else         __builtin_amdgcn_s_wait_tensorcnt(0);  // drain
    }
    asm volatile("" ::: "memory");   // TDM wrote Alds behind the compiler's back
    __syncthreads();

    // A frag: lane<16 holds K {0..7,16..23}, lane>=16 holds K {8..15,24..31}
    v16bf afr[2], bfr[4];
    #pragma unroll
    for (int mi = 0; mi < 2; ++mi) {
      int r  = wm*32 + mi*16 + lrow;
      int kb = lhi * 8;
      FragCvt f;
      f.u[0] = *(const uint4*)&Alds[cur][r*LSTRIDE + kb];
      f.u[1] = *(const uint4*)&Alds[cur][r*LSTRIDE + kb + 16];
      afr[mi] = f.v;
    }
    // B frag: lane<16 holds K {0..7,16..23} of column lrow, lane>=16 holds K {8..15,24..31}
    #pragma unroll
    for (int ni = 0; ni < 4; ++ni) {
      int n  = wn*64 + ni*16 + lrow;
      int kb = lhi * 8;
      FragCvt f;
      f.u[0] = *(const uint4*)&Blds[cur][n*LSTRIDE + kb];
      f.u[1] = *(const uint4*)&Blds[cur][n*LSTRIDE + kb + 16];
      bfr[ni] = f.v;
    }
    #pragma unroll
    for (int mi = 0; mi < 2; ++mi)
      #pragma unroll
      for (int ni = 0; ni < 4; ++ni)
        acc[mi][ni] = __builtin_amdgcn_wmma_f32_16x16x32_bf16(
            false, afr[mi], false, bfr[ni], (short)0, acc[mi][ni], false, false);
    __syncthreads();   // readers of `cur` done before it is restaged
  }

  #pragma unroll
  for (int mi = 0; mi < 2; ++mi)
    #pragma unroll
    for (int ni = 0; ni < 4; ++ni)
      #pragma unroll
      for (int i = 0; i < 8; ++i)
        C[(size_t)(rowBase + mi*16 + lhi*8 + i) * N + colBase + ni*16 + lrow] = acc[mi][ni][i];
}

// ---------------------------------------------------------------------------
// Embedding gather: h[n,:] = emb[ids[n],:]
// ---------------------------------------------------------------------------
__global__ __launch_bounds__(256) void gather_kernel(
    const int* __restrict__ ids, const float* __restrict__ emb, float* __restrict__ h)
{
  int n = blockIdx.x, tid = threadIdx.x;
  size_t src = (size_t)ids[n] * DIM;
  #pragma unroll
  for (int i = 0; i < 4; ++i)
    h[(size_t)n*DIM + tid + 256*i] = emb[src + tid + 256*i];
}

// ---------------------------------------------------------------------------
// RMSNorm, then either split into relu(+/-) bf16 pair or plain bf16
// ---------------------------------------------------------------------------
__global__ __launch_bounds__(256) void rmsnorm_kernel(
    const float* __restrict__ x, const float* __restrict__ w,
    __bf16* __restrict__ outp, __bf16* __restrict__ outn)
{
  __shared__ float red[256];
  int n = blockIdx.x, tid = threadIdx.x;
  const float* row = x + (size_t)n * DIM;
  float v[4]; float ss = 0.f;
  #pragma unroll
  for (int i = 0; i < 4; ++i) { v[i] = row[tid + 256*i]; ss += v[i]*v[i]; }
  red[tid] = ss; __syncthreads();
  for (int s = 128; s > 0; s >>= 1) { if (tid < s) red[tid] += red[tid + s]; __syncthreads(); }
  float scale = rsqrtf(red[0] * (1.0f / DIM) + 1e-5f);
  #pragma unroll
  for (int i = 0; i < 4; ++i) {
    int j = tid + 256*i;
    float y = v[i] * scale * w[j];
    if (outn) {
      outp[(size_t)n*DIM + j] = (__bf16)fmaxf(y, 0.f);
      outn[(size_t)n*DIM + j] = (__bf16)fmaxf(-y, 0.f);
    } else {
      outp[(size_t)n*DIM + j] = (__bf16)y;
    }
  }
}

// ---------------------------------------------------------------------------
// Per-token spiking attention. One block per token. IF membranes live in
// registers across the 10 timesteps; Q(16x64)Kt(8x64) and S@V in LDS/VALU.
// Output stored as dim = d*16 + h (matches the reference transpose+reshape),
// pre-scaled by 2*SCALE/T = 0.025 (KV heads are tiled twice).
// ---------------------------------------------------------------------------
__global__ __launch_bounds__(256) void spike_attn_kernel(
    const float* __restrict__ cqp, const float* __restrict__ cqn,
    const float* __restrict__ ckp, const float* __restrict__ ckn,
    const float* __restrict__ cvp, const float* __restrict__ cvn,
    __bf16* __restrict__ xa)
{
  __shared__ float qs[1024], ks[512], vs[512], Sm[128];
  int n = blockIdx.x, tid = threadIdx.x;
  float cq0[4], cq1[4], ck0[2], ck1[2], cv0[2], cv1[2];
  float vq0[4] = {0,0,0,0}, vq1[4] = {0,0,0,0};
  float vk0[2] = {0,0}, vk1[2] = {0,0}, vv0[2] = {0,0}, vv1[2] = {0,0};
  #pragma unroll
  for (int i = 0; i < 4; ++i) {
    int j = tid + 256*i;
    cq0[i] = cqp[(size_t)n*1024 + j]; cq1[i] = cqn[(size_t)n*1024 + j];
  }
  #pragma unroll
  for (int i = 0; i < 2; ++i) {
    int j = tid + 256*i;
    ck0[i] = ckp[(size_t)n*512 + j]; ck1[i] = ckn[(size_t)n*512 + j];
    cv0[i] = cvp[(size_t)n*512 + j]; cv1[i] = cvn[(size_t)n*512 + j];
  }
  float acc[4] = {0,0,0,0};
  for (int t = 0; t < TSTEPS; ++t) {
    #pragma unroll
    for (int i = 0; i < 4; ++i) {
      vq0[i] += cq0[i]; float s0 = vq0[i] >= 1.f ? 1.f : 0.f; vq0[i] *= (1.f - s0);
      vq1[i] += cq1[i]; float s1 = vq1[i] >= 1.f ? 1.f : 0.f; vq1[i] *= (1.f - s1);
      qs[tid + 256*i] = s0 - s1;
    }
    #pragma unroll
    for (int i = 0; i < 2; ++i) {
      vk0[i] += ck0[i]; float s0 = vk0[i] >= 1.f ? 1.f : 0.f; vk0[i] *= (1.f - s0);
      vk1[i] += ck1[i]; float s1 = vk1[i] >= 1.f ? 1.f : 0.f; vk1[i] *= (1.f - s1);
      ks[tid + 256*i] = s0 - s1;
      vv0[i] += cv0[i]; float t0 = vv0[i] >= 1.f ? 1.f : 0.f; vv0[i] *= (1.f - t0);
      vv1[i] += cv1[i]; float t1 = vv1[i] >= 1.f ? 1.f : 0.f; vv1[i] *= (1.f - t1);
      vs[tid + 256*i] = t0 - t1;
    }
    __syncthreads();
    if (tid < 128) {                       // S[h][g] = q_h . k_g
      int h = tid >> 3, g = tid & 7;
      float s = 0.f;
      for (int d = 0; d < 64; ++d) s += qs[h*64 + d] * ks[g*64 + d];
      Sm[tid] = s;
    }
    __syncthreads();
    #pragma unroll
    for (int i = 0; i < 4; ++i) {          // xa[h][d] += sum_g S[h][g]*v[g][d]
      int o = tid + 256*i, h = o >> 6, d = o & 63;
      float s = 0.f;
      #pragma unroll
      for (int g = 0; g < 8; ++g) s += Sm[h*8 + g] * vs[g*64 + d];
      acc[i] += s;
    }
    __syncthreads();
  }
  #pragma unroll
  for (int i = 0; i < 4; ++i) {
    int o = tid + 256*i, h = o >> 6, d = o & 63;
    xa[(size_t)n*1024 + d*16 + h] = (__bf16)(acc[i] * 0.025f);  // 2*SCALE/T
  }
}

// ---------------------------------------------------------------------------
// FFN: g = rate(xp@W1) * (rate(xp@W3) - rate(xn@W3)), rate = fires/10
// ---------------------------------------------------------------------------
__device__ __forceinline__ float spike_rate(float c) {
  float v = 0.f, cnt = 0.f;
  #pragma unroll
  for (int t = 0; t < TSTEPS; ++t) {
    v += c; float s = v >= 1.f ? 1.f : 0.f; cnt += s; v *= (1.f - s);
  }
  return cnt * 0.1f;
}

__global__ __launch_bounds__(256) void ffn_rate_kernel(
    const float* __restrict__ c1, const float* __restrict__ c3p,
    const float* __restrict__ c3n, __bf16* __restrict__ g)
{
  size_t i = (size_t)blockIdx.x * 256 + threadIdx.x;
  g[i] = (__bf16)(spike_rate(c1[i]) * (spike_rate(c3p[i]) - spike_rate(c3n[i])));
}

// ---------------------------------------------------------------------------
extern "C" void kernel_launch(void* const* d_in, const int* in_sizes, int n_in,
                              void* d_out, int out_size, void* d_ws, size_t ws_size,
                              hipStream_t stream) {
  const int*   ids  = (const int*)d_in[0];
  const float* emb  = (const float*)d_in[1];
  const float* anw  = (const float*)d_in[2];
  const float* wq   = (const float*)d_in[3];
  const float* wk   = (const float*)d_in[4];
  const float* wv   = (const float*)d_in[5];
  const float* wo   = (const float*)d_in[6];
  const float* fnw  = (const float*)d_in[7];
  const float* w1   = (const float*)d_in[8];
  const float* w2   = (const float*)d_in[9];
  const float* w3   = (const float*)d_in[10];
  const float* finw = (const float*)d_in[11];
  const float* wout = (const float*)d_in[12];
  float* out = (float*)d_out;

  char* ws = (char*)d_ws;
  size_t off = 0;
  auto take = [&](size_t bytes) -> char* {
    char* p = ws + off; off += (bytes + 255) & ~(size_t)255; return p;
  };
  float*  h  = (float*) take((size_t)NTOK * DIM * 4);
  __bf16* xp = (__bf16*)take((size_t)NTOK * DIM * 2);
  __bf16* xn = (__bf16*)take((size_t)NTOK * DIM * 2);
  char* region = take((size_t)56 * 1024 * 1024);      // attn/ffn scratch (overlapped)
  // attention-phase views (18 MB)
  float*  cqp = (float*)(region);
  float*  cqn = (float*)(region + ((size_t)4  << 20));
  float*  ckp = (float*)(region + ((size_t)8  << 20));
  float*  ckn = (float*)(region + ((size_t)10 << 20));
  float*  cvp = (float*)(region + ((size_t)12 << 20));
  float*  cvn = (float*)(region + ((size_t)14 << 20));
  __bf16* xa  = (__bf16*)(region + ((size_t)16 << 20));
  // ffn-phase views (56 MB, overlap attention views)
  float*  c1   = (float*)(region);
  float*  c3p  = (float*)(region + ((size_t)16 << 20));
  float*  c3n  = (float*)(region + ((size_t)32 << 20));
  __bf16* gact = (__bf16*)(region + ((size_t)48 << 20));

  dim3 blk(256);
  gather_kernel<<<NTOK, blk, 0, stream>>>(ids, emb, h);

  for (int l = 0; l < 2; ++l) {
    const float* wq_l = wq + (size_t)l * 1024 * 1024;
    const float* wk_l = wk + (size_t)l * 1024 * 512;
    const float* wv_l = wv + (size_t)l * 1024 * 512;
    const float* wo_l = wo + (size_t)l * 1024 * 1024;
    const float* w1_l = w1 + (size_t)l * 1024 * 4096;
    const float* w2_l = w2 + (size_t)l * 4096 * 1024;
    const float* w3_l = w3 + (size_t)l * 1024 * 4096;

    // ---- attention branch ----
    rmsnorm_kernel<<<NTOK, blk, 0, stream>>>(h, anw + l*DIM, xp, xn);
    gemm_bf16<<<dim3(8, 8), blk, 0, stream>>>(xp, wq_l, cqp, 1024, 1024, 1024, 0);
    gemm_bf16<<<dim3(8, 8), blk, 0, stream>>>(xn, wq_l, cqn, 1024, 1024, 1024, 0);
    gemm_bf16<<<dim3(4, 8), blk, 0, stream>>>(xp, wk_l, ckp, 1024,  512, 1024, 0);
    gemm_bf16<<<dim3(4, 8), blk, 0, stream>>>(xn, wk_l, ckn, 1024,  512, 1024, 0);
    gemm_bf16<<<dim3(4, 8), blk, 0, stream>>>(xp, wv_l, cvp, 1024,  512, 1024, 0);
    gemm_bf16<<<dim3(4, 8), blk, 0, stream>>>(xn, wv_l, cvn, 1024,  512, 1024, 0);
    spike_attn_kernel<<<NTOK, blk, 0, stream>>>(cqp, cqn, ckp, ckn, cvp, cvn, xa);
    gemm_bf16<<<dim3(8, 8), blk, 0, stream>>>(xa, wo_l, h, 1024, 1024, 1024, 1); // h += xa@Wo

    // ---- ffn branch ----
    rmsnorm_kernel<<<NTOK, blk, 0, stream>>>(h, fnw + l*DIM, xp, xn);
    gemm_bf16<<<dim3(32, 8), blk, 0, stream>>>(xp, w1_l, c1,  1024, 4096, 1024, 0);
    gemm_bf16<<<dim3(32, 8), blk, 0, stream>>>(xp, w3_l, c3p, 1024, 4096, 1024, 0);
    gemm_bf16<<<dim3(32, 8), blk, 0, stream>>>(xn, w3_l, c3n, 1024, 4096, 1024, 0);
    ffn_rate_kernel<<<(NTOK * HIDDEN) / 256, blk, 0, stream>>>(c1, c3p, c3n, gact);
    gemm_bf16<<<dim3(8, 8), blk, 0, stream>>>(gact, w2_l, h, 1024, 1024, 4096, 1); // h += g@W2
  }

  // ---- final norm + logits ----
  rmsnorm_kernel<<<NTOK, blk, 0, stream>>>(h, finw, xp, nullptr);
  gemm_bf16<<<dim3(250, 8), blk, 0, stream>>>(xp, wout, out, 1024, 32000, 1024, 0);
}
